// LinearAttention_4690104287949
// MI455X (gfx1250) — matrix operand order repaired and checked by reading, not compiled, not run on hardware
//
#include <hip/hip_runtime.h>
#include <stdint.h>

typedef __attribute__((ext_vector_type(16))) _Float16 v16h;
typedef __attribute__((ext_vector_type(8)))  float    v8f;

#define N_ 4
#define L_ 4096
#define S_ 4096
#define H_ 8
#define D_ 64
#define V_ 64
#define EPS_ 1e-6f

#define SCHUNKS 8
#define SCHUNK (S_ / SCHUNKS)   // 512 rows of S per block
#define NSTEP  (SCHUNK / 32)    // 16 stages of 32 rows
#define KT_PITCH 34             // halfs; 68B rows -> dword aligned, conflict-free strides
#define KVT_PITCH 66            // halfs; 132B rows -> dword aligned

__device__ __forceinline__ float qk_map(float x) {
    // elu(x)+1 : x>0 ? x+1 : exp(x)
    return (x > 0.0f) ? (x + 1.0f) : __expf(x);
}

__global__ void zero_ws_kernel(float* __restrict__ ws, int n) {
    int i = blockIdx.x * blockDim.x + threadIdx.x;
    if (i < n) ws[i] = 0.0f;
}

// ---------------------------------------------------------------------------
// Phase 1: KV[n,h,d,v] = sum_s K[n,s,h,d]*v[n,s,h,v]; Ksum[n,h,d] = sum_s K.
// One block per (n,h,S-chunk). 8 waves, each wave owns two 16x16 KV tiles
// sharing one d-block (one A fill feeds two WMMAs). Global loads for stage
// i+1 are issued between the barriers of stage i (software pipeline), and
// Ksum partials are register-resident (one LDS fold at the end).
// ---------------------------------------------------------------------------
__global__ __launch_bounds__(256) void kv_kernel(
    const float* __restrict__ k, const float* __restrict__ v,
    const unsigned char* __restrict__ kvm,
    float* __restrict__ kvws, float* __restrict__ ksumws)
{
    __shared__ _Float16 Kt[D_ * KT_PITCH];   // Kt[d][s]  (transposed, f16 after map)
    __shared__ _Float16 Vt[V_ * KT_PITCH];   // Vt[v][s]
    __shared__ float    ksumLds[D_];

    const int b     = blockIdx.x;
    const int chunk = b & (SCHUNKS - 1);
    const int nh    = b >> 3;                // log2(SCHUNKS) == 3
    const int n     = nh / H_;
    const int h     = nh % H_;
    const int t     = threadIdx.x;
    const int lane  = t & 31;
    const int wid   = t >> 5;
    const int m     = lane & 15;
    const int kbA   = (lane < 16) ? 0 : 8;   // A operand: K in groups of 16 halfs (8 dwords)
    const int kbB   = (lane < 16) ? 0 : 8;   // B operand: K in groups of 16 halfs (8 dwords)

    if (t < D_) ksumLds[t] = 0.0f;

    // Fixed staging slice per thread: one s-row, 8 consecutive d.
    const int srow = t >> 3;                 // 0..31
    const int d0   = (t & 7) * 8;            // 0,8,...,56

    const int ti0 = wid * 2;
    const int td  = ti0 >> 2;                // both tiles share the d-block
    const int tv0 = ti0 & 3;
    v8f acc0 = {};
    v8f acc1 = {};

    const uint32_t* arow  = (const uint32_t*)&Kt[(td * 16 + m) * KT_PITCH];
    const uint32_t* brow0 = (const uint32_t*)&Vt[((tv0 + 0) * 16 + m) * KT_PITCH];
    const uint32_t* brow1 = (const uint32_t*)&Vt[((tv0 + 1) * 16 + m) * KT_PITCH];

    const int s0 = chunk * SCHUNK;
    float kreg[8], vreg[8], msk;
    float ksr[8] = {0.0f, 0.0f, 0.0f, 0.0f, 0.0f, 0.0f, 0.0f, 0.0f};

    auto issue_loads = [&](int sbase) {
        const size_t gi = ((size_t)((n * S_ + sbase + srow) * H_ + h)) * 64 + d0;
        *(float4*)&kreg[0] = *(const float4*)&k[gi];
        *(float4*)&kreg[4] = *(const float4*)&k[gi + 4];
        *(float4*)&vreg[0] = *(const float4*)&v[gi];
        *(float4*)&vreg[4] = *(const float4*)&v[gi + 4];
        msk = kvm[n * S_ + sbase + srow] ? 1.0f : 0.0f;
    };

    issue_loads(s0);

    for (int step = 0; step < NSTEP; ++step) {
        if (step) __syncthreads();           // previous stage fully consumed
        #pragma unroll
        for (int j = 0; j < 8; ++j) {        // map + mask + f16 + transposed store
            const float kf = qk_map(kreg[j]) * msk;
            ksr[j] += kf;
            Kt[(d0 + j) * KT_PITCH + srow] = (_Float16)kf;
            Vt[(d0 + j) * KT_PITCH + srow] = (_Float16)(vreg[j] * msk);
        }
        __syncthreads();
        if (step + 1 < NSTEP) issue_loads(s0 + (step + 1) * 32);  // hide behind WMMA

        v16h a, b0, b1;
        uint32_t* a32 = (uint32_t*)&a;
        uint32_t* b0p = (uint32_t*)&b0;
        uint32_t* b1p = (uint32_t*)&b1;
        #pragma unroll
        for (int r = 0; r < 4; ++r) {        // A: lanes<16 K=0..15, lanes>=16 K=16..31
            a32[r]     = arow[kbA + r];
            a32[4 + r] = arow[kbA + 4 + r];
        }
        #pragma unroll
        for (int r = 0; r < 8; ++r) {        // B: lanes<16 K=0..15, lanes>=16 K=16..31
            b0p[r] = brow0[kbB + r];
            b1p[r] = brow1[kbB + r];
        }
        acc0 = __builtin_amdgcn_wmma_f32_16x16x32_f16(false, a, false, b0, (short)0, acc0, false, false);
        acc1 = __builtin_amdgcn_wmma_f32_16x16x32_f16(false, a, false, b1, (short)0, acc1, false, false);
    }

    // Fold Ksum register partials (32 threads share each d-range).
    #pragma unroll
    for (int j = 0; j < 8; ++j) atomicAdd(&ksumLds[d0 + j], ksr[j]);

    // Fold partial KV tiles.  C layout: VGPR r -> M=r (lanes 0-15) / M=8+r (lanes 16-31).
    float* kvbase = kvws + (size_t)(n * H_ + h) * D_ * V_;
    const int rowoff = (lane < 16) ? 0 : 8;
    #pragma unroll
    for (int r = 0; r < 8; ++r) {
        const int row = td * 16 + rowoff + r;
        atomicAdd(&kvbase[row * V_ + (tv0 + 0) * 16 + m], acc0[r]);
        atomicAdd(&kvbase[row * V_ + (tv0 + 1) * 16 + m], acc1[r]);
    }
    __syncthreads();
    if (t < D_) atomicAdd(&ksumws[(n * H_ + h) * D_ + t], ksumLds[t]);
}

// ---------------------------------------------------------------------------
// Phase 2: out[n,l,h,v] = (Q[l,:]·KV) / (Q[l,:]·Ksum + eps).
// One block per (n,h,128 q-rows). Each wave owns a 16-row stripe:
// 4 output tiles x 2 K-steps = 8 WMMAs. A comes straight from global q
// (d contiguous per row), B from LDS-staged transposed f16 KV.
// ---------------------------------------------------------------------------
__global__ __launch_bounds__(256) void out_kernel(
    const float* __restrict__ q, const unsigned char* __restrict__ qm,
    const float* __restrict__ kvws, const float* __restrict__ ksumws,
    float* __restrict__ out)
{
    __shared__ _Float16 KVt[V_ * KVT_PITCH];  // KVt[v][d] (transposed f16)
    __shared__ float    ksumLds[D_];
    __shared__ float    zs[8 * 16];           // per-wave per-row Z denominators

    const int b    = blockIdx.x;
    const int lb   = b & 31;                  // L_/128 == 32
    const int nh   = b >> 5;
    const int n    = nh / H_;
    const int h    = nh % H_;
    const int t    = threadIdx.x;
    const int lane = t & 31;
    const int wid  = t >> 5;
    const int m    = lane & 15;
    const int kbA  = (lane < 16) ? 0 : 8;     // A: dword offset (2*kbA halfs) within 32-wide K window
    const int kbB  = (lane < 16) ? 0 : 8;     // B: dword offset (16-half groups)

    // Stage 64x64 KV: each thread owns row d = t>>2, 16 consecutive v (4 x b128).
    const float* kvbase = kvws + (size_t)(n * H_ + h) * D_ * V_;
    {
        const int dd  = t >> 2;
        const int vv0 = (t & 3) * 16;
        float kvreg[16];
        const float* src = kvbase + dd * V_ + vv0;
        *(float4*)&kvreg[0]  = *(const float4*)&src[0];
        *(float4*)&kvreg[4]  = *(const float4*)&src[4];
        *(float4*)&kvreg[8]  = *(const float4*)&src[8];
        *(float4*)&kvreg[12] = *(const float4*)&src[12];
        #pragma unroll
        for (int j = 0; j < 16; ++j)
            KVt[(vv0 + j) * KVT_PITCH + dd] = (_Float16)kvreg[j];
    }
    if (t < D_)  ksumLds[t] = ksumws[(n * H_ + h) * D_ + t];
    if (t < 128) zs[t] = 0.0f;
    __syncthreads();

    const int l0   = lb * 128 + wid * 16;
    const int lrow = l0 + m;
    const float* qrow = q + ((size_t)((n * L_ + lrow) * H_ + h)) * 64;
    const float  msk  = qm[n * L_ + lrow] ? 1.0f : 0.0f;

    v8f acc[4] = {};
    float zpart = 0.0f;
    #pragma unroll
    for (int kstep = 0; kstep < 2; ++kstep) {
        const int w = kstep * 32;
        v16h a;
        #pragma unroll
        for (int j = 0; j < 8; ++j) {
            const int d1 = w + 2 * kbA + j;        // halfs 0..7 of A regs
            float qf = qk_map(qrow[d1]) * msk;
            zpart += qf * ksumLds[d1];
            a[j] = (_Float16)qf;
            const int d2 = w + 2 * kbA + 8 + j;    // halfs 8..15 of A regs
            qf = qk_map(qrow[d2]) * msk;
            zpart += qf * ksumLds[d2];
            a[8 + j] = (_Float16)qf;
        }
        #pragma unroll
        for (int tv = 0; tv < 4; ++tv) {
            v16h bb;
            uint32_t* bp = (uint32_t*)&bb;
            const uint32_t* br = (const uint32_t*)&KVt[(tv * 16 + m) * KVT_PITCH];
            #pragma unroll
            for (int r = 0; r < 8; ++r) bp[r] = br[kstep * 16 + kbB + r];
            acc[tv] = __builtin_amdgcn_wmma_f32_16x16x32_f16(false, a, false, bb, (short)0, acc[tv], false, false);
        }
    }
    atomicAdd(&zs[wid * 16 + m], zpart);      // lanes m and m+16 each hold half the dot
    __syncthreads();

    const int rowoff = (lane < 16) ? 0 : 8;
    #pragma unroll
    for (int r = 0; r < 8; ++r) {
        const int l = l0 + rowoff + r;
        const float z = 1.0f / (zs[wid * 16 + rowoff + r] + EPS_);
        float* orow = out + ((size_t)((n * L_ + l) * H_ + h)) * 64;
        #pragma unroll
        for (int tv = 0; tv < 4; ++tv)
            orow[tv * 16 + m] = acc[tv][r] * z;
    }
}

extern "C" void kernel_launch(void* const* d_in, const int* in_sizes, int n_in,
                              void* d_out, int out_size, void* d_ws, size_t ws_size,
                              hipStream_t stream) {
    (void)in_sizes; (void)n_in; (void)out_size; (void)ws_size;
    const float* q  = (const float*)d_in[0];
    const float* k  = (const float*)d_in[1];
    const float* v  = (const float*)d_in[2];
    const unsigned char* qmask  = (const unsigned char*)d_in[3];
    const unsigned char* kvmask = (const unsigned char*)d_in[4];
    float* out = (float*)d_out;

    float* kvws   = (float*)d_ws;                          // N*H*D*V f32
    float* ksumws = kvws + (size_t)N_ * H_ * D_ * V_;      // N*H*D f32

    const int ztotal = N_ * H_ * (D_ * V_ + D_);
    hipLaunchKernelGGL(zero_ws_kernel, dim3((ztotal + 255) / 256), dim3(256), 0, stream,
                       kvws, ztotal);
    hipLaunchKernelGGL(kv_kernel, dim3(N_ * H_ * SCHUNKS), dim3(256), 0, stream,
                       k, v, kvmask, kvws, ksumws);
    hipLaunchKernelGGL(out_kernel, dim3(N_ * H_ * (L_ / 128)), dim3(256), 0, stream,
                       q, qmask, kvws, ksumws, out);
}
